// RNN_35957466202559
// MI455X (gfx1250) — compile-verified, matrix operand order checked by reading, not run on hardware
//
#include <hip/hip_runtime.h>
#include <math.h>

// ---------------- problem constants ----------------
#define EMB    512
#define HID    1024
#define SEQ    128
#define BATCH  64
#define VOCAB  10000
#define LAYERS 2

typedef __attribute__((ext_vector_type(16))) __bf16    v16bf;
typedef __attribute__((ext_vector_type(8)))  float     v8f;
typedef __attribute__((ext_vector_type(8)))  unsigned  v8u;

// ---------------- bf16 helpers ----------------
static __device__ __forceinline__ unsigned short f32_to_bf16(float f) {
    unsigned x = __builtin_bit_cast(unsigned, f);
    unsigned r = (x + 0x7FFFu + ((x >> 16) & 1u)) >> 16;
    return (unsigned short)r;
}
static __device__ __forceinline__ float bf16_to_f32(unsigned short u) {
    return __builtin_bit_cast(float, ((unsigned)u) << 16);
}

// K-pair base for 16-bit A/B WMMA fragments (CDNA5 ISA 7.12.2):
// vgpr v, lane half h: k0 = 2*(v&3) + 16*(v>>2) + 8*h   (holds elements k0,k0+1)
static __device__ __forceinline__ int kpair_base(int v, int half) {
    return ((v & 3) << 1) + ((v >> 2) << 4) + (half << 3);
}

// load one 16x32 bf16 fragment (A-layout; B uses same layout on transposed weights)
static __device__ __forceinline__ v16bf load_frag(const unsigned* __restrict__ rowp,
                                                  int kb, int half) {
    v8u u;
#pragma unroll
    for (int v = 0; v < 8; ++v) {
        u[v] = rowp[(kb + kpair_base(v, half)) >> 1];
    }
    return __builtin_bit_cast(v16bf, u);
}

// ---------------- conversion kernels ----------------
__global__ void cvt_scale_kernel(const float* __restrict__ src,
                                 unsigned short* __restrict__ dst,
                                 int n, float scale) {
    int i = blockIdx.x * blockDim.x + threadIdx.x;
    if (i < n) dst[i] = f32_to_bf16(src[i] * scale);
}

// fp32 (K,N) row-major -> bf16 (N,K) row-major (so B fragments load contiguously)
__global__ void transpose_cvt_kernel(const float* __restrict__ src,
                                     unsigned short* __restrict__ dst,
                                     int K, int N) {
    int i = blockIdx.x * blockDim.x + threadIdx.x;
    if (i < K * N) {
        int k = i / N, n = i - k * N;
        dst[(size_t)n * K + k] = f32_to_bf16(src[i]);
    }
}

__global__ void init_hidden_kernel(const float* __restrict__ hidden,
                                   unsigned short* __restrict__ h0_slot0,
                                   unsigned short* __restrict__ h1_slot0) {
    int i = blockIdx.x * blockDim.x + threadIdx.x;
    if (i < BATCH * HID) {
        h0_slot0[i] = f32_to_bf16(hidden[i]);
        h1_slot0[i] = f32_to_bf16(hidden[BATCH * HID + i]);
    }
}

// ---------------- recurrent step: Hout = tanh(A@WaT^T + Hprev@WhT^T + bias) ----
// A: layer0 -> gathered emb rows (tok != nullptr, KA=EMB); layer1 -> h0_new (KA=HID)
// 256 16x16 tiles, 1 wave each: grid 64 x 128 threads. Latency-bound phase;
// 4 waves of a block share tn -> weight fragments reused through WGP$.
__global__ void rnn_step_kernel(const unsigned short* __restrict__ Abase,
                                const int* __restrict__ tok, int KA,
                                const unsigned short* __restrict__ WaT,
                                const unsigned short* __restrict__ Hprev,
                                const unsigned short* __restrict__ WhT,
                                const float* __restrict__ bias,
                                unsigned short* __restrict__ Hout) {
    const int wid  = threadIdx.x >> 5;
    const int tile = blockIdx.x * 4 + wid;          // 0..255
    const int tm   = tile & 3;                      // 4 row tiles (batch)
    const int tn   = tile >> 2;                     // 64 col tiles
    const int lane = threadIdx.x & 31;
    const int half = lane >> 4;
    const int mn   = lane & 15;
    const int row  = tm * 16 + mn;                  // batch row for A fragment
    const int ncol = tn * 16 + mn;                  // output col for B fragment

    const unsigned* Arow = tok
        ? (const unsigned*)(Abase + (size_t)tok[row] * KA)
        : (const unsigned*)(Abase + (size_t)row * KA);
    const unsigned* Ba   = (const unsigned*)(WaT + (size_t)ncol * KA);
    const unsigned* Hrow = (const unsigned*)(Hprev + (size_t)row * HID);
    const unsigned* Bh   = (const unsigned*)(WhT + (size_t)ncol * HID);

    v8f acc = {};
    for (int kb = 0; kb < KA; kb += 32) {
        acc = __builtin_amdgcn_wmma_f32_16x16x32_bf16(
            false, load_frag(Arow, kb, half),
            false, load_frag(Ba, kb, half),
            (short)0, acc, false, false);
    }
    for (int kb = 0; kb < HID; kb += 32) {
        acc = __builtin_amdgcn_wmma_f32_16x16x32_bf16(
            false, load_frag(Hrow, kb, half),
            false, load_frag(Bh, kb, half),
            (short)0, acc, false, false);
    }

#pragma unroll
    for (int r = 0; r < 8; ++r) {                   // C/D layout: m = r + 8*half
        int m = tm * 16 + r + 8 * half;
        int n = tn * 16 + mn;
        float val = tanhf(acc[r] + bias[n]);
        Hout[(size_t)m * HID + n] = f32_to_bf16(val);
    }
}

// ---------------- output projection: (8192 x 10000) = A @ WoutT^T + b_out -----
// Register-blocked: each wave computes a 32x80 block (2 m-tiles x 5 n-tiles):
// per K-step 7 fragment loads, 10 WMMAs. All 8 waves of a workgroup share the
// same nb (B fragments loaded once per block, 8-way WGP$ reuse) and take 8
// consecutive mb -> ~61 FLOP per byte of L2 traffic; compute-limited at the
// WMMA pipes. Grid: 32 m-groups x 125 n-blocks = 4000 blocks x 8 waves.
#define MB 2
#define NB 5
__global__ void logits_kernel(const unsigned short* __restrict__ A,
                              const unsigned short* __restrict__ WoutT,
                              const float* __restrict__ bout,
                              float* __restrict__ out) {
    const int wid  = threadIdx.x >> 5;
    const int nb   = blockIdx.x % 125;              // shared by all 8 waves
    const int mb   = (blockIdx.x / 125) * 8 + wid;  // 0..255, distinct per wave
    const int lane = threadIdx.x & 31;
    const int half = lane >> 4;
    const int mn   = lane & 15;

    const unsigned* Arow[MB];
    const unsigned* Brow[NB];
#pragma unroll
    for (int i = 0; i < MB; ++i)
        Arow[i] = (const unsigned*)(A + (size_t)(mb * (MB * 16) + i * 16 + mn) * HID);
#pragma unroll
    for (int j = 0; j < NB; ++j)
        Brow[j] = (const unsigned*)(WoutT + (size_t)(nb * (NB * 16) + j * 16 + mn) * HID);

    v8f acc[MB][NB];
#pragma unroll
    for (int i = 0; i < MB; ++i)
#pragma unroll
        for (int j = 0; j < NB; ++j) acc[i][j] = (v8f){};

    for (int kb = 0; kb < HID; kb += 32) {
        v16bf a[MB], b[NB];
#pragma unroll
        for (int i = 0; i < MB; ++i) a[i] = load_frag(Arow[i], kb, half);
#pragma unroll
        for (int j = 0; j < NB; ++j) b[j] = load_frag(Brow[j], kb, half);
#pragma unroll
        for (int i = 0; i < MB; ++i)
#pragma unroll
            for (int j = 0; j < NB; ++j)
                acc[i][j] = __builtin_amdgcn_wmma_f32_16x16x32_bf16(
                    false, a[i], false, b[j], (short)0, acc[i][j], false, false);
    }

#pragma unroll
    for (int i = 0; i < MB; ++i) {
#pragma unroll
        for (int j = 0; j < NB; ++j) {
            int ncol = nb * (NB * 16) + j * 16 + mn;
            float bj = bout[ncol];
#pragma unroll
            for (int r = 0; r < 8; ++r) {
                int m = mb * (MB * 16) + i * 16 + r + 8 * half;
                out[(size_t)m * VOCAB + ncol] = acc[i][j][r] + bj;
            }
        }
    }
}

// ---------------- final hidden: bf16 states -> fp32 tail of d_out -------------
__global__ void finalize_hidden_kernel(const unsigned short* __restrict__ h0_last,
                                       const unsigned short* __restrict__ h1_last,
                                       float* __restrict__ out) {
    int i = blockIdx.x * blockDim.x + threadIdx.x;
    if (i < BATCH * HID) {
        out[i]               = bf16_to_f32(h0_last[i]);
        out[BATCH * HID + i] = bf16_to_f32(h1_last[i]);
    }
}

// ---------------- workspace layout (bytes) ----------------
#define OFF_EMB   ((size_t)0)                                  // VOCAB*EMB bf16 (pre-scaled)
#define OFF_WX0   (OFF_EMB  + (size_t)VOCAB * EMB * 2)         // HIDxEMB bf16 (transposed)
#define OFF_WX1   (OFF_WX0  + (size_t)HID * EMB * 2)           // HIDxHID bf16
#define OFF_WH0   (OFF_WX1  + (size_t)HID * HID * 2)           // HIDxHID bf16
#define OFF_WH1   (OFF_WH0  + (size_t)HID * HID * 2)           // HIDxHID bf16
#define OFF_WOUT  (OFF_WH1  + (size_t)HID * HID * 2)           // VOCABxHID bf16
#define OFF_H0    (OFF_WOUT + (size_t)VOCAB * HID * 2)         // 2 slots of BATCHxHID bf16
#define OFF_H1ALL (OFF_H0   + (size_t)2 * BATCH * HID * 2)     // (SEQ+1) slots of BATCHxHID bf16

extern "C" void kernel_launch(void* const* d_in, const int* in_sizes, int n_in,
                              void* d_out, int out_size, void* d_ws, size_t ws_size,
                              hipStream_t stream) {
    const int*   inputs = (const int*)  d_in[0];
    const float* hidden = (const float*)d_in[1];
    const float* emb    = (const float*)d_in[2];
    const float* W_x0   = (const float*)d_in[3];
    const float* W_x1   = (const float*)d_in[4];
    const float* W_h    = (const float*)d_in[5];
    const float* b_h    = (const float*)d_in[6];
    const float* W_out  = (const float*)d_in[7];
    const float* b_out  = (const float*)d_in[8];
    float* out = (float*)d_out;

    char* ws = (char*)d_ws;
    unsigned short* emb_bf = (unsigned short*)(ws + OFF_EMB);
    unsigned short* wx0T   = (unsigned short*)(ws + OFF_WX0);
    unsigned short* wx1T   = (unsigned short*)(ws + OFF_WX1);
    unsigned short* wh0T   = (unsigned short*)(ws + OFF_WH0);
    unsigned short* wh1T   = (unsigned short*)(ws + OFF_WH1);
    unsigned short* woutT  = (unsigned short*)(ws + OFF_WOUT);
    unsigned short* h0buf  = (unsigned short*)(ws + OFF_H0);     // 2 slots
    unsigned short* h1all  = (unsigned short*)(ws + OFF_H1ALL);  // SEQ+1 slots

    const int TPB = 256;
    const int ST  = BATCH * HID;  // 65536 elements per state slot

    // weight conversion (read-once; all bf16 weights fit in 192MB L2 afterwards)
    cvt_scale_kernel<<<(VOCAB * EMB + TPB - 1) / TPB, TPB, 0, stream>>>(
        emb, emb_bf, VOCAB * EMB, 22.62741699796952f /* sqrt(512) */);
    transpose_cvt_kernel<<<(EMB * HID + TPB - 1) / TPB, TPB, 0, stream>>>(W_x0, wx0T, EMB, HID);
    transpose_cvt_kernel<<<(HID * HID + TPB - 1) / TPB, TPB, 0, stream>>>(W_x1, wx1T, HID, HID);
    transpose_cvt_kernel<<<(HID * HID + TPB - 1) / TPB, TPB, 0, stream>>>(W_h,             wh0T, HID, HID);
    transpose_cvt_kernel<<<(HID * HID + TPB - 1) / TPB, TPB, 0, stream>>>(W_h + HID * HID, wh1T, HID, HID);
    transpose_cvt_kernel<<<(HID * VOCAB + TPB - 1) / TPB, TPB, 0, stream>>>(W_out, woutT, HID, VOCAB);
    init_hidden_kernel<<<(ST + TPB - 1) / TPB, TPB, 0, stream>>>(hidden, h0buf, h1all);

    // sequential recurrence: 2 small WMMA kernels per timestep (graph amortizes)
    for (int t = 0; t < SEQ; ++t) {
        unsigned short* h0_prev = h0buf + (size_t)(t & 1) * ST;
        unsigned short* h0_new  = h0buf + (size_t)((t + 1) & 1) * ST;
        rnn_step_kernel<<<64, 128, 0, stream>>>(
            emb_bf, inputs + t * BATCH, EMB, wx0T, h0_prev, wh0T, b_h, h0_new);
        rnn_step_kernel<<<64, 128, 0, stream>>>(
            h0_new, nullptr, HID, wx1T, h1all + (size_t)t * ST, wh1T, b_h + HID,
            h1all + (size_t)(t + 1) * ST);
    }

    // big parallel output GEMM: (8192 x 10000) over K=1024, register-blocked bf16 WMMA
    logits_kernel<<<4000, 256, 0, stream>>>(h1all + ST, woutT, b_out, out);

    // final hidden states (h0 parity after 128 steps -> slot 0)
    finalize_hidden_kernel<<<(ST + TPB - 1) / TPB, TPB, 0, stream>>>(
        h0buf, h1all + (size_t)SEQ * ST, out + (size_t)SEQ * BATCH * VOCAB);
}
